// GinNet_64991445123389
// MI455X (gfx1250) — matrix operand order repaired and checked by reading, not compiled, not run on hardware
//
#include <hip/hip_runtime.h>
#include <math.h>

// ---------------- problem constants ----------------
constexpr int N_NODES  = 100000;
constexpr int N_EDGES  = 1600000;
constexpr int F_IN     = 64;
constexpr int H        = 200;
constexpr int N_GRAPHS = 512;

constexpr int NT_TILES = 13;          // 13 * 16 = 208 >= 200
// LDS row padding (bank-conflict-free b128 fragment loads)
constexpr int KPAD_L1  = 72;          // for K=64
constexpr int KPAD_L2  = 232;         // for K=224 (200 padded)

// ---------------- workspace layout (bytes) ----------------
constexpr size_t OFF_AGG  = 0;                       // 100000*200*4 = 80,000,000 (reused L1/L2)
constexpr size_t OFF_H1   = 80000000;                // 100000*200*4
constexpr size_t OFF_PACK = 160000000;               // packed bf16 weights
constexpr size_t SZ1      = 26 * 512;                // elements per W1a matrix (2 kc * 13 nt * 32 lanes * 16)/16... = tiles*512
constexpr size_t SZ2      = 91 * 512;                // 7 kc * 13 nt tiles
constexpr size_t OFF_BIAS = OFF_PACK + (2 * SZ1 + 6 * SZ2) * 2;  // 160,612,352
constexpr size_t OFF_POOL = OFF_BIAS + 8 * 208 * 4;              // 160,619,008
// pool_sum: 512*200*4 = 409600 ; pool_maxkey: 409600

// ---------------- WMMA types ----------------
typedef __attribute__((ext_vector_type(16))) __bf16 v16bf;
typedef __attribute__((ext_vector_type(8)))  __bf16 v8bf;
typedef __attribute__((ext_vector_type(8)))  float  v8f;

union FragU { v16bf v; v8bf h[2]; };

template<int KPAD>
__device__ __forceinline__ v16bf load_a_frag(const __bf16* base, int M, int k0) {
  FragU u;
  u.h[0] = *(const v8bf*)(base + M * KPAD + k0);
  u.h[1] = *(const v8bf*)(base + M * KPAD + k0 + 16);
  return u.v;
}

__device__ __forceinline__ v16bf load_b_frag(const __bf16* __restrict__ p, int tile, int lane) {
  const v8bf* q = (const v8bf*)(p + (((size_t)tile * 32 + lane) << 4));
  FragU u;
  u.h[0] = q[0];
  u.h[1] = q[1];
  return u.v;
}

__device__ __forceinline__ unsigned fmax_key(float f) {
  unsigned b = __float_as_uint(f);
  return (b & 0x80000000u) ? ~b : (b | 0x80000000u);
}

// ---------------- small prep kernels ----------------
__global__ void prep_params(const float* b1a, const float* b1b, const float* b2a, const float* b2b,
                            const float* g1, const float* be1, const float* m1, const float* v1,
                            const float* g2, const float* be2, const float* m2, const float* v2,
                            float* out) {
  int n = threadIdx.x;
  if (n >= 208) return;
  bool ok = n < H;
  out[0 * 208 + n] = ok ? b1a[n] : 0.f;
  out[1 * 208 + n] = ok ? b1b[n] : 0.f;
  out[2 * 208 + n] = ok ? b2a[n] : 0.f;
  out[3 * 208 + n] = ok ? b2b[n] : 0.f;
  float s1 = ok ? g1[n] * rsqrtf(v1[n] + 1e-5f) : 0.f;
  float s2 = ok ? g2[n] * rsqrtf(v2[n] + 1e-5f) : 0.f;
  out[4 * 208 + n] = s1;
  out[5 * 208 + n] = ok ? (be1[n] - m1[n] * s1) : 0.f;
  out[6 * 208 + n] = s2;
  out[7 * 208 + n] = ok ? (be2[n] - m2[n] * s2) : 0.f;
}

// Pack W[K x 200] (row-major) into per-(kc,nt) WMMA B fragments, split hi/lo bf16.
// Fragment layout per tile: 32 lanes x 16 values; lane L: N = nt*16 + (L&15),
// K = kc*32 + (L>>4)*8 + (i<8 ? i : i+8)  (i = 0..15)  -- matches 16-bit B 32x16 layout.
__global__ void pack_weights(const float* __restrict__ W, int K, int KC,
                             __bf16* __restrict__ dHi, __bf16* __restrict__ dLo) {
  int gid = blockIdx.x * blockDim.x + threadIdx.x;
  int total = KC * NT_TILES * 32;
  if (gid >= total) return;
  int lane = gid & 31;
  int tile = gid >> 5;
  int kc = tile / NT_TILES, nt = tile % NT_TILES;
  int n = nt * 16 + (lane & 15);
  int ksBase = kc * 32 + (lane >> 4) * 8;
  size_t o = (size_t)gid * 16;
#pragma unroll
  for (int i = 0; i < 16; ++i) {
    int ks = ksBase + (i < 8 ? i : i + 8);
    float w = (ks < K && n < H) ? W[(size_t)ks * H + n] : 0.f;
    __bf16 hi = (__bf16)w;
    dHi[o + i] = hi;
    dLo[o + i] = (__bf16)(w - (float)hi);
  }
}

__global__ void zero_f32(float* __restrict__ p, long long n) {
  long long i = ((long long)blockIdx.x * blockDim.x + threadIdx.x) * 4;
  long long stride = (long long)gridDim.x * blockDim.x * 4;
  for (; i < n; i += stride) *(float4*)(p + i) = make_float4(0.f, 0.f, 0.f, 0.f);
}

__global__ void init_pool(float* __restrict__ psum, unsigned* __restrict__ pmax, int n) {
  int i = blockIdx.x * blockDim.x + threadIdx.x;
  if (i < n) { psum[i] = 0.f; pmax[i] = 0x007FFFFFu; }  // encodes -inf
}

// agg[dst] += x[src], vectorized float4 per thread; F4 = F/4
__global__ void scatter_add(const float* __restrict__ x, const int* __restrict__ src,
                            const int* __restrict__ dst, float* __restrict__ agg,
                            int F4, long long total) {
  long long t = (long long)blockIdx.x * blockDim.x + threadIdx.x;
  if (t >= total) return;
  int e = (int)(t / F4);
  int j = (int)(t % F4);
  int s = src[e], d = dst[e];
  float4 v = ((const float4*)x)[(size_t)s * F4 + j];
  float* a = agg + ((size_t)d * F4 + j) * 4;
  atomicAdd(a + 0, v.x);
  atomicAdd(a + 1, v.y);
  atomicAdd(a + 2, v.z);
  atomicAdd(a + 3, v.w);
}

// ---------------- fused GIN layer ----------------
// h0 = (1+eps)*x + agg ; t = relu(h0 @ Wa + ba) ; y = bn(relu(t @ Wb + bb))
// POOL=false: write y to hout[N x 200] ; POOL=true: atomically pool y into sum/max[G x 200].
template<int KPAD1, int KC1, bool POOL>
__global__ __launch_bounds__(128) void gin_layer(
    const float* __restrict__ xin, const float* __restrict__ agg,
    const float* __restrict__ epsp, int FIN,
    const __bf16* __restrict__ WaHi, const __bf16* __restrict__ WaLo, const float* __restrict__ ba_p,
    const __bf16* __restrict__ WbHi, const __bf16* __restrict__ WbLo, const float* __restrict__ bb_p,
    const float* __restrict__ scale_p, const float* __restrict__ shift_p,
    float* __restrict__ hout,
    const int* __restrict__ batch,
    float* __restrict__ pool_sum, unsigned* __restrict__ pool_maxkey) {
  constexpr int KPAD2 = KPAD_L2;  // second GEMM always K=200 -> 224 padded
  constexpr int KC2 = 7;

  __shared__ __bf16 AHi1[4][16][KPAD1];
  __shared__ __bf16 ALo1[4][16][KPAD1];
  __shared__ __bf16 AHi2[4][16][KPAD2];
  __shared__ __bf16 ALo2[4][16][KPAD2];

  const int lane = threadIdx.x & 31;
  const int wave = threadIdx.x >> 5;
  const int tileBase = (blockIdx.x * 4 + wave) * 16;
  const float e1 = 1.0f + epsp[0];

  // ---- stage split-bf16 input tile into LDS ----
  const int KLOG1 = KC1 * 32;
  for (int idx = lane; idx < 16 * KLOG1; idx += 32) {
    int r = idx / KLOG1, c = idx % KLOG1;
    int node = tileBase + r;
    float v = 0.f;
    if (node < N_NODES && c < FIN)
      v = e1 * xin[(size_t)node * FIN + c] + agg[(size_t)node * FIN + c];
    __bf16 hi = (__bf16)v;
    AHi1[wave][r][c] = hi;
    ALo1[wave][r][c] = (__bf16)(v - (float)hi);
  }

  const int M = lane & 15;
  const int kb = (lane >> 4) * 8;
  const int colN = lane & 15;
  const int rowOff = (lane >> 4) * 8;

  // ---- GEMM1: [16 x KLOG1] @ Wa -> 13 tiles of 16x16, split-bf16 (3 wmma / chunk) ----
  v8f acc[NT_TILES];
#pragma unroll
  for (int nt = 0; nt < NT_TILES; ++nt) acc[nt] = {};
  for (int kc = 0; kc < KC1; ++kc) {
    v16bf ahi = load_a_frag<KPAD1>(&AHi1[wave][0][0], M, kc * 32 + kb);
    v16bf alo = load_a_frag<KPAD1>(&ALo1[wave][0][0], M, kc * 32 + kb);
#pragma unroll
    for (int nt = 0; nt < NT_TILES; ++nt) {
      int tile = kc * NT_TILES + nt;
      v16bf bhi = load_b_frag(WaHi, tile, lane);
      v16bf blo = load_b_frag(WaLo, tile, lane);
      acc[nt] = __builtin_amdgcn_wmma_f32_16x16x32_bf16(false, ahi, false, bhi, (short)0, acc[nt], false, false);
      acc[nt] = __builtin_amdgcn_wmma_f32_16x16x32_bf16(false, ahi, false, blo, (short)0, acc[nt], false, false);
      acc[nt] = __builtin_amdgcn_wmma_f32_16x16x32_bf16(false, alo, false, bhi, (short)0, acc[nt], false, false);
    }
  }

  // ---- epilogue 1: bias + relu, re-split into LDS as A for GEMM2 ----
#pragma unroll
  for (int nt = 0; nt < NT_TILES; ++nt) {
    int n = nt * 16 + colN;
    float b = ba_p[n];
#pragma unroll
    for (int r = 0; r < 8; ++r) {
      float v = acc[nt][r] + b;
      v = v > 0.f ? v : 0.f;
      __bf16 hi = (__bf16)v;
      AHi2[wave][rowOff + r][n] = hi;
      ALo2[wave][rowOff + r][n] = (__bf16)(v - (float)hi);
    }
  }
  // zero pad cols 208..223 (read by GEMM2's K loop)
  for (int idx = lane; idx < 16 * 16; idx += 32) {
    int r = idx >> 4, c = 208 + (idx & 15);
    AHi2[wave][r][c] = (__bf16)0.f;
    ALo2[wave][r][c] = (__bf16)0.f;
  }

  // ---- GEMM2: [16 x 224] @ Wb ----
  v8f acc2[NT_TILES];
#pragma unroll
  for (int nt = 0; nt < NT_TILES; ++nt) acc2[nt] = {};
  for (int kc = 0; kc < KC2; ++kc) {
    v16bf ahi = load_a_frag<KPAD2>(&AHi2[wave][0][0], M, kc * 32 + kb);
    v16bf alo = load_a_frag<KPAD2>(&ALo2[wave][0][0], M, kc * 32 + kb);
#pragma unroll
    for (int nt = 0; nt < NT_TILES; ++nt) {
      int tile = kc * NT_TILES + nt;
      v16bf bhi = load_b_frag(WbHi, tile, lane);
      v16bf blo = load_b_frag(WbLo, tile, lane);
      acc2[nt] = __builtin_amdgcn_wmma_f32_16x16x32_bf16(false, ahi, false, bhi, (short)0, acc2[nt], false, false);
      acc2[nt] = __builtin_amdgcn_wmma_f32_16x16x32_bf16(false, ahi, false, blo, (short)0, acc2[nt], false, false);
      acc2[nt] = __builtin_amdgcn_wmma_f32_16x16x32_bf16(false, alo, false, bhi, (short)0, acc2[nt], false, false);
    }
  }

  // ---- epilogue 2: bias + relu + BN affine, then store or pool ----
#pragma unroll
  for (int nt = 0; nt < NT_TILES; ++nt) {
    int n = nt * 16 + colN;
    if (n < H) {
      float b = bb_p[n], sc = scale_p[n], sh = shift_p[n];
#pragma unroll
      for (int r = 0; r < 8; ++r) {
        int node = tileBase + rowOff + r;
        if (node < N_NODES) {
          float v = acc2[nt][r] + b;
          v = v > 0.f ? v : 0.f;
          v = v * sc + sh;
          if (!POOL) {
            hout[(size_t)node * H + n] = v;
          } else {
            int g = batch[node];
            atomicAdd(&pool_sum[g * H + n], v);
            atomicMax(&pool_maxkey[g * H + n], fmax_key(v));
          }
        }
      }
    }
  }
}

// ---------------- FC + log_softmax ----------------
__global__ void fc_logsoftmax(const float* __restrict__ psum, const unsigned* __restrict__ pmax,
                              const float* __restrict__ Wfc, const float* __restrict__ bfc,
                              float* __restrict__ out) {
  int g = blockIdx.x * blockDim.x + threadIdx.x;
  if (g >= N_GRAPHS) return;
  float z0 = bfc[0], z1 = bfc[1];
  for (int k = 0; k < H; ++k) {
    float s = psum[g * H + k];
    z0 += s * Wfc[k * 2 + 0];
    z1 += s * Wfc[k * 2 + 1];
  }
  for (int k = 0; k < H; ++k) {
    unsigned kk = pmax[g * H + k];
    unsigned b = (kk & 0x80000000u) ? (kk & 0x7FFFFFFFu) : ~kk;
    float m = __uint_as_float(b);
    z0 += m * Wfc[(H + k) * 2 + 0];
    z1 += m * Wfc[(H + k) * 2 + 1];
  }
  float mx = fmaxf(z0, z1);
  float l = mx + logf(expf(z0 - mx) + expf(z1 - mx));
  out[g * 2 + 0] = z0 - l;
  out[g * 2 + 1] = z1 - l;
}

// ---------------- host launcher ----------------
extern "C" void kernel_launch(void* const* d_in, const int* in_sizes, int n_in,
                              void* d_out, int out_size, void* d_ws, size_t ws_size,
                              hipStream_t stream) {
  const float* x     = (const float*)d_in[0];
  const int*   ei    = (const int*)d_in[1];   // [2][E], row0 = src, row1 = dst
  const int*   batch = (const int*)d_in[2];
  const float* eps1  = (const float*)d_in[3];
  const float* W1a   = (const float*)d_in[4];
  const float* b1a   = (const float*)d_in[5];
  const float* W1b   = (const float*)d_in[6];
  const float* b1b   = (const float*)d_in[7];
  const float* g1    = (const float*)d_in[8];
  const float* be1   = (const float*)d_in[9];
  const float* m1    = (const float*)d_in[10];
  const float* v1    = (const float*)d_in[11];
  const float* eps2  = (const float*)d_in[12];
  const float* W2a   = (const float*)d_in[13];
  const float* b2a   = (const float*)d_in[14];
  const float* W2b   = (const float*)d_in[15];
  const float* b2b   = (const float*)d_in[16];
  const float* g2    = (const float*)d_in[17];
  const float* be2   = (const float*)d_in[18];
  const float* m2    = (const float*)d_in[19];
  const float* v2    = (const float*)d_in[20];
  const float* Wfc   = (const float*)d_in[21];
  const float* bfc   = (const float*)d_in[22];

  char* ws = (char*)d_ws;
  float*    agg   = (float*)(ws + OFF_AGG);
  float*    h1    = (float*)(ws + OFF_H1);
  __bf16*   pk    = (__bf16*)(ws + OFF_PACK);
  __bf16*   W1aHi = pk;
  __bf16*   W1aLo = pk + SZ1;
  __bf16*   W1bHi = pk + 2 * SZ1;
  __bf16*   W1bLo = pk + 2 * SZ1 + SZ2;
  __bf16*   W2aHi = pk + 2 * SZ1 + 2 * SZ2;
  __bf16*   W2aLo = pk + 2 * SZ1 + 3 * SZ2;
  __bf16*   W2bHi = pk + 2 * SZ1 + 4 * SZ2;
  __bf16*   W2bLo = pk + 2 * SZ1 + 5 * SZ2;
  float*    prm   = (float*)(ws + OFF_BIAS);
  float*    b1a_p = prm + 0 * 208;
  float*    b1b_p = prm + 1 * 208;
  float*    b2a_p = prm + 2 * 208;
  float*    b2b_p = prm + 3 * 208;
  float*    sc1   = prm + 4 * 208;
  float*    sh1   = prm + 5 * 208;
  float*    sc2   = prm + 6 * 208;
  float*    sh2   = prm + 7 * 208;
  float*    psum  = (float*)(ws + OFF_POOL);
  unsigned* pmax  = (unsigned*)(ws + OFF_POOL + (size_t)N_GRAPHS * H * 4);

  const int* src = ei;
  const int* dst = ei + N_EDGES;

  // params + packed weights
  prep_params<<<1, 256, 0, stream>>>(b1a, b1b, b2a, b2b, g1, be1, m1, v1, g2, be2, m2, v2, prm);
  pack_weights<<<4, 256, 0, stream>>>(W1a, 64, 2, W1aHi, W1aLo);
  pack_weights<<<12, 256, 0, stream>>>(W1b, 200, 7, W1bHi, W1bLo);
  pack_weights<<<12, 256, 0, stream>>>(W2a, 200, 7, W2aHi, W2aLo);
  pack_weights<<<12, 256, 0, stream>>>(W2b, 200, 7, W2bHi, W2bLo);
  init_pool<<<(N_GRAPHS * H + 255) / 256, 256, 0, stream>>>(psum, pmax, N_GRAPHS * H);

  // layer 1: aggregate (F=64) then fused MLP+BN -> h1
  zero_f32<<<2048, 256, 0, stream>>>(agg, (long long)N_NODES * F_IN);
  {
    long long total = (long long)N_EDGES * (F_IN / 4);
    scatter_add<<<(unsigned)((total + 255) / 256), 256, 0, stream>>>(x, src, dst, agg, F_IN / 4, total);
  }
  gin_layer<KPAD_L1, 2, false><<<1563, 128, 0, stream>>>(
      x, agg, eps1, F_IN, W1aHi, W1aLo, b1a_p, W1bHi, W1bLo, b1b_p, sc1, sh1,
      h1, nullptr, nullptr, nullptr);

  // layer 2: aggregate (F=200) then fused MLP+BN + pooling atomics
  zero_f32<<<4096, 256, 0, stream>>>(agg, (long long)N_NODES * H);
  {
    long long total = (long long)N_EDGES * (H / 4);
    scatter_add<<<(unsigned)((total + 255) / 256), 256, 0, stream>>>(h1, src, dst, agg, H / 4, total);
  }
  gin_layer<KPAD_L2, 7, true><<<1563, 128, 0, stream>>>(
      h1, agg, eps2, H, W2aHi, W2aLo, b2a_p, W2bHi, W2bLo, b2b_p, sc2, sh2,
      nullptr, batch, psum, pmax);

  // readout
  fc_logsoftmax<<<2, 256, 0, stream>>>(psum, pmax, Wfc, bfc, (float*)d_out);
}
